// NTM_65197603553303
// MI455X (gfx1250) — matrix operand order repaired and checked by reading, not compiled, not run on hardware
//
#include <hip/hip_runtime.h>
#include <hip/hip_bf16.h>

// ---------------------------------------------------------------------------
// NTM scan for MI455X (gfx1250, wave32, WMMA).
//
// Persistent-grid design: 16 workgroups (one per WGP), 256 threads (8 waves).
//  - WGP k owns gate columns {g*512 + 32k .. +32 | g=i,f,g,o} = 8 WMMA n-tiles.
//  - Weight B-fragments (bf16, K padded 616->640) live in VGPRs: 20 frags/wave
//    (160 VGPRs), loaded once -- zero steady-state weight traffic.
//  - Per step: build A panel in LDS from [x_t | r | pad | h_prev], 20 WMMAs
//    per wave (A loads double-buffered against WMMA issue), LSTM update of the
//    local h/c slice, global barrier, then WGP b computes heads + NTM memory
//    update for batch b (memory resident in LDS), writes r, global barrier.
// ---------------------------------------------------------------------------

typedef __attribute__((ext_vector_type(16))) __bf16 v16bf;
typedef __attribute__((ext_vector_type(8)))  float  v8f;

#define B_    16
#define T_    1024
#define IN_   64
#define OUT_  64
#define H_    512
#define NMEM  128
#define MMEM  40
#define KPAD  640        // 128 (padded input+read) + 512 (h)
#define NCHUNK 20        // KPAD / 32
#define NWG   16
#define NTILES 8         // 128 gate cols per WGP / 16
#define EPSF  1e-8f

// workspace layout (bytes); needs ~2.7 MB
#define FRAG_ELEMS ((size_t)NWG * NTILES * NCHUNK * 32 * 16)  // 1,310,720 bf16
#define WS_FRAG_OFF ((size_t)0)
#define WS_H_OFF    (FRAG_ELEMS * 2)                          // 2 slabs of B*H f32
#define WS_R_OFF    (WS_H_OFF + (size_t)2 * B_ * H_ * 4)
#define WS_CNT_OFF  (WS_R_OFF + (size_t)B_ * MMEM * 4)

__device__ __forceinline__ float sigm(float x) { return 1.f / (1.f + expf(-x)); }

// grid-wide barrier over NWG resident workgroups (device-scope atomics).
// Tight acquire-poll: 16 pollers on one L2 line is negligible; lowest wakeup
// latency matters more than poll traffic for this recurrence.
__device__ __forceinline__ void gsync(unsigned* cnt, unsigned target) {
  __syncthreads();
  if (threadIdx.x == 0) {
    __threadfence();
    __hip_atomic_fetch_add(cnt, 1u, __ATOMIC_RELEASE, __HIP_MEMORY_SCOPE_AGENT);
    while (__hip_atomic_load(cnt, __ATOMIC_ACQUIRE, __HIP_MEMORY_SCOPE_AGENT) < target) {
    }
  }
  __syncthreads();
}

// ---------------------------------------------------------------------------
// Prep: pack [W_ih | 0pad | W_hh] into WMMA B-fragment layout (bf16) and init
// h slab0 = h0, r = read0, barrier counter = 0.
// B-fragment (32x16 bf16), this code's k order (same as A), lane L half h:  K = (L>=16 ? 16 : 0) + h, N = L%16.
// ---------------------------------------------------------------------------
__global__ void ntm_pack(const float* __restrict__ W_ih, const float* __restrict__ W_hh,
                         const float* __restrict__ h0,  const float* __restrict__ read0,
                         __bf16* __restrict__ frag, float* __restrict__ hbuf,
                         float* __restrict__ rbuf, unsigned* __restrict__ cnt)
{
  size_t idx = (size_t)blockIdx.x * blockDim.x + threadIdx.x;
  if (idx < FRAG_ELEMS) {
    int h    = (int)(idx & 15);
    int lane = (int)((idx >> 4) & 31);
    size_t rest = idx >> 9;
    int c = (int)(rest % NCHUNK);
    rest /= NCHUNK;
    int w = (int)(rest % NTILES);
    int k = (int)(rest / NTILES);
    int n  = (w >> 1) * 512 + 32 * k + (w & 1) * 16 + (lane & 15);
    int kk = 32 * c + ((lane >= 16) ? 16 : 0) + h;
    float v;
    if (kk < 128) v = (kk < (IN_ + MMEM)) ? W_ih[(size_t)n * (IN_ + MMEM) + kk] : 0.f;
    else          v = W_hh[(size_t)n * H_ + (kk - 128)];
    frag[idx] = (__bf16)v;
  }
  if (idx < (size_t)B_ * H_)    hbuf[idx] = h0[idx];     // slab 0 = h0
  if (idx < (size_t)B_ * MMEM)  rbuf[idx] = read0[idx];
  if (idx == 0)                 *cnt = 0u;
}

// ---------------------------------------------------------------------------
// Persistent scan kernel.
// ---------------------------------------------------------------------------
__global__ __launch_bounds__(256, 1) void ntm_scan(
    const float* __restrict__ x,    const float* __restrict__ c0,
    const float* __restrict__ mem0,
    const float* __restrict__ b_ih, const float* __restrict__ b_hh,
    const float* __restrict__ W_fc, const float* __restrict__ b_fc,
    const float* __restrict__ W_e,  const float* __restrict__ b_e,
    const float* __restrict__ W_a,  const float* __restrict__ b_a,
    const float* __restrict__ W_k,  const float* __restrict__ b_k,
    const float* __restrict__ W_b,  const float* __restrict__ b_b,
    const __bf16* __restrict__ frag, float* __restrict__ hbuf,
    float* __restrict__ rbuf, unsigned* __restrict__ cnt,
    float* __restrict__ out)
{
  __shared__ __align__(32) __bf16 Apanel[NCHUNK * 512];   // 20 KB: A fragments
  __shared__ float gatesBuf[128 * 16];                    // 8 KB: [col128][batch]
  __shared__ float cs[512];                               // c slice [batch][32]
  __shared__ __align__(16) float hb[H_];                  // h_t of this batch
  __shared__ float memb[NMEM * MMEM];                     // 20 KB NTM memory
  __shared__ float keyv[MMEM], ev[MMEM], av[MMEM];
  __shared__ float wv[NMEM];
  __shared__ float red[2];
  __shared__ float betav;

  const int tid  = threadIdx.x;
  const int lane = tid & 31;
  const int wave = tid >> 5;
  const int blk  = blockIdx.x;

  // ---- load resident weight B-fragments into VGPRs (wave w owns n-tile w) --
  v16bf bfr[NCHUNK];
  {
    const __bf16* base = frag + ((size_t)(blk * NTILES + wave) * NCHUNK) * 512;
#pragma unroll
    for (int c = 0; c < NCHUNK; ++c)
      bfr[c] = *(const v16bf*)(base + (size_t)c * 512 + lane * 16);
  }

  // ---- init persistent state ----------------------------------------------
  for (int i = tid; i < 512; i += 256)
    cs[i] = c0[(i >> 5) * H_ + 32 * blk + (i & 31)];
  for (int i = tid; i < NMEM * MMEM; i += 256)
    memb[i] = mem0[(size_t)blk * NMEM * MMEM + i];

  unsigned bar = 0;

  for (int t = 0; t < T_; ++t) {
    const float* hprev = hbuf + (size_t)(t & 1) * (B_ * H_);
    float*       hnew  = hbuf + (size_t)((t + 1) & 1) * (B_ * H_);

    // ---- build A panel: inp = [x_t | r | 0pad | h_prev], 16 x 640 bf16 ----
    // A-fragment (16x32 bf16), this code's k order (same as B), lane L half h: M=L%16,
    //   K = 16*(L>=16) + h
    for (int i = tid; i < B_ * KPAD; i += 256) {
      int m = i / KPAD, kk = i % KPAD;
      float v;
      if (kk < IN_)             v = x[((size_t)m * T_ + t) * IN_ + kk];
      else if (kk < IN_ + MMEM) v = rbuf[m * MMEM + (kk - IN_)];
      else if (kk < 128)        v = 0.f;
      else                      v = hprev[m * H_ + (kk - 128)];
      int c  = kk >> 5, kl = kk & 31;
      int ln = m + ((kl >> 4) & 1) * 16;
      int hh = kl & 15;
      Apanel[c * 512 + ln * 16 + hh] = (__bf16)v;
    }
    __syncthreads();

    // ---- gates GEMM: one 16x16 tile per wave, K = 640 (20 WMMAs) ----------
    // A loads double-buffered: issue chunk c+1's ds_load before WMMA c so the
    // backend can overlap LDS latency with matrix issue (partial dscnt waits).
    v8f acc = {0.f, 0.f, 0.f, 0.f, 0.f, 0.f, 0.f, 0.f};
    {
      v16bf acur = *(const v16bf*)(Apanel + lane * 16);
#pragma unroll
      for (int c = 0; c < NCHUNK; ++c) {
        v16bf anext = acur;
        if (c + 1 < NCHUNK)
          anext = *(const v16bf*)(Apanel + (c + 1) * 512 + lane * 16);
        acc = __builtin_amdgcn_wmma_f32_16x16x32_bf16(false, acur, false, bfr[c],
                                                      (short)0, acc, false, false);
        acur = anext;
      }
    }
    {
      // C/D layout: lane L, float f -> M = f + 8*(L>=16), N = L%16
      int ncol = wave * 16 + (lane & 15);
      int mb   = (lane >> 4) * 8;
#pragma unroll
      for (int f = 0; f < 8; ++f) gatesBuf[ncol * 16 + mb + f] = acc[f];
    }
    __syncthreads();

    // ---- LSTM update for h indices [32*blk, 32*blk+32), all 16 batches ----
    for (int i = tid; i < 512; i += 256) {
      int m = i >> 5, hoff = i & 31;
      int nb = 32 * blk + hoff;
      float gi = gatesBuf[(hoff)      * 16 + m] + b_ih[nb]        + b_hh[nb];
      float gf = gatesBuf[(32 + hoff) * 16 + m] + b_ih[512 + nb]  + b_hh[512 + nb];
      float gg = gatesBuf[(64 + hoff) * 16 + m] + b_ih[1024 + nb] + b_hh[1024 + nb];
      float go = gatesBuf[(96 + hoff) * 16 + m] + b_ih[1536 + nb] + b_hh[1536 + nb];
      gi = sigm(gi); gf = sigm(gf); gg = tanhf(gg); go = sigm(go);
      float cn = gf * cs[i] + gi * gg;
      cs[i] = cn;
      hnew[m * H_ + nb] = go * tanhf(cn);
    }
    gsync(cnt, bar += NWG);   // h_t complete everywhere

    // ---- heads for batch b = blk ------------------------------------------
    for (int i = tid; i < H_; i += 256) hb[i] = hnew[blk * H_ + i];
    __syncthreads();
    if (tid < 185) {
      const float* wr; float bias;
      int j = tid;
      if (j < 64)       { wr = W_fc + (size_t)j * H_;         bias = b_fc[j]; }
      else if (j < 104) { wr = W_k  + (size_t)(j - 64) * H_;  bias = b_k[j - 64]; }
      else if (j < 144) { wr = W_e  + (size_t)(j - 104) * H_; bias = b_e[j - 104]; }
      else if (j < 184) { wr = W_a  + (size_t)(j - 144) * H_; bias = b_a[j - 144]; }
      else              { wr = W_b;                           bias = b_b[0]; }
      float d = bias;
      const float4* w4 = (const float4*)wr;
      const float4* h4 = (const float4*)hb;
      for (int q = 0; q < H_ / 4; ++q) {
        float4 a4 = w4[q]; float4 b4 = h4[q];
        d += a4.x * b4.x + a4.y * b4.y + a4.z * b4.z + a4.w * b4.w;
      }
      if (j < 64)       out[((size_t)blk * T_ + t) * OUT_ + j] = tanhf(d);
      else if (j < 104) keyv[j - 64]  = tanhf(d);
      else if (j < 144) ev[j - 104]   = sigm(d);
      else if (j < 184) av[j - 144]   = tanhf(d);
      else              betav = ((d > 20.f) ? d : log1pf(expf(d))) + EPSF;
    }
    __syncthreads();

    // ---- content addressing: cosine sim + softmax -------------------------
    if (tid == 0) {
      float s = 0.f;
      for (int q = 0; q < MMEM; ++q) s += keyv[q] * keyv[q];
      red[0] = sqrtf(s);                         // ||key||
    }
    __syncthreads();
    if (tid < NMEM) {
      float dot = 0.f, nn = 0.f;
      for (int q = 0; q < MMEM; ++q) {
        float v = memb[tid * MMEM + q];
        dot += v * keyv[q]; nn += v * v;
      }
      float sim = dot / ((sqrtf(nn) + EPSF) * (red[0] + EPSF));
      wv[tid] = betav * sim;
    }
    __syncthreads();
    if (tid == 0) {
      float mx = wv[0];
      for (int q = 1; q < NMEM; ++q) mx = fmaxf(mx, wv[q]);
      red[1] = mx;
    }
    __syncthreads();
    if (tid < NMEM) wv[tid] = expf(wv[tid] - red[1]);
    __syncthreads();
    if (tid == 0) {
      float s = 0.f;
      for (int q = 0; q < NMEM; ++q) s += wv[q];
      red[0] = 1.f / s;
    }
    __syncthreads();
    if (tid < NMEM) wv[tid] *= red[0];
    __syncthreads();

    // ---- erase/add write, then read r = w . mem ---------------------------
    for (int i = tid; i < NMEM * MMEM; i += 256) {
      int n = i / MMEM, q = i % MMEM;
      float wn = wv[n];
      memb[i] = memb[i] * (1.f - wn * ev[q]) + wn * av[q];
    }
    __syncthreads();
    if (tid < MMEM) {
      float s = 0.f;
      for (int n = 0; n < NMEM; ++n) s += wv[n] * memb[n * MMEM + tid];
      rbuf[blk * MMEM + tid] = s;
    }
    gsync(cnt, bar += NWG);   // r_t complete everywhere
  }

  // ---- final h and c outputs ----------------------------------------------
  const float* hfin = hbuf + (size_t)(T_ & 1) * (B_ * H_);
  const size_t h_off = (size_t)B_ * T_ * OUT_;
  const size_t c_off = h_off + (size_t)B_ * H_;
  for (int i = tid; i < 512; i += 256) {
    int m = i >> 5, hoff = i & 31;
    out[h_off + (size_t)m * H_ + 32 * blk + hoff] = hfin[m * H_ + 32 * blk + hoff];
    out[c_off + (size_t)m * H_ + 32 * blk + hoff] = cs[i];
  }
}

// ---------------------------------------------------------------------------
extern "C" void kernel_launch(void* const* d_in, const int* in_sizes, int n_in,
                              void* d_out, int out_size, void* d_ws, size_t ws_size,
                              hipStream_t stream)
{
  (void)in_sizes; (void)n_in; (void)out_size; (void)ws_size; // needs ~2.7 MB ws
  const float* x    = (const float*)d_in[0];
  const float* h0   = (const float*)d_in[1];
  const float* c0   = (const float*)d_in[2];
  const float* mem0 = (const float*)d_in[3];
  const float* r0   = (const float*)d_in[4];
  const float* W_ih = (const float*)d_in[5];
  const float* b_ih = (const float*)d_in[6];
  const float* W_hh = (const float*)d_in[7];
  const float* b_hh = (const float*)d_in[8];
  const float* W_fc = (const float*)d_in[9];
  const float* b_fc = (const float*)d_in[10];
  const float* W_e  = (const float*)d_in[11];
  const float* b_e  = (const float*)d_in[12];
  const float* W_a  = (const float*)d_in[13];
  const float* b_a  = (const float*)d_in[14];
  const float* W_k  = (const float*)d_in[15];
  const float* b_k  = (const float*)d_in[16];
  const float* W_b  = (const float*)d_in[17];
  const float* b_b  = (const float*)d_in[18];

  char* ws = (char*)d_ws;
  __bf16*   frag = (__bf16*)(ws + WS_FRAG_OFF);
  float*    hbuf = (float*)(ws + WS_H_OFF);
  float*    rbuf = (float*)(ws + WS_R_OFF);
  unsigned* cnt  = (unsigned*)(ws + WS_CNT_OFF);

  const int threads = 256;
  const int blocks  = (int)((FRAG_ELEMS + threads - 1) / threads);
  ntm_pack<<<blocks, threads, 0, stream>>>(W_ih, W_hh, h0, r0, frag, hbuf, rbuf, cnt);
  ntm_scan<<<NWG, 256, 0, stream>>>(x, c0, mem0, b_ih, b_hh, W_fc, b_fc,
                                    W_e, b_e, W_a, b_a, W_k, b_k, W_b, b_b,
                                    frag, hbuf, rbuf, cnt, (float*)d_out);
}
